// HGRNAttention_69398081569487
// MI455X (gfx1250) — matrix operand order repaired and checked by reading, not compiled, not run on hardware
//
#include <hip/hip_runtime.h>
#include <hip/hip_bf16.h>
#include <stdint.h>

// ---------------------------------------------------------------------------
// Types for CDNA5 WMMA (wave32, 16x16x32 bf16 -> f32)
// ---------------------------------------------------------------------------
typedef __attribute__((ext_vector_type(16))) __bf16 v16bf;
typedef __attribute__((ext_vector_type(8)))  __bf16 v8bf;
typedef __attribute__((ext_vector_type(8)))  float  v8f;
typedef __attribute__((ext_vector_type(4)))  float  v4f;

__device__ __forceinline__ __bf16 f2bf(float f) {
    unsigned u = __builtin_bit_cast(unsigned, f);
    unsigned r = u + 0x7FFFu + ((u >> 16) & 1u);   // round-to-nearest-even
    unsigned short s = (unsigned short)(r >> 16);
    return __builtin_bit_cast(__bf16, s);
}

__device__ __forceinline__ float sigmoidf_(float x) {
    return 1.0f / (1.0f + __expf(-x));
}

// ---------------------------------------------------------------------------
// fp32 -> bf16 conversion, 8 elements / thread
// ---------------------------------------------------------------------------
__global__ __launch_bounds__(256)
void cvt_bf16_kernel(const float* __restrict__ src, __bf16* __restrict__ dst, size_t n) {
    size_t i = ((size_t)blockIdx.x * blockDim.x + threadIdx.x) * 8;
    if (i >= n) return;
    v4f a = *(const v4f*)(src + i);
    v4f b = *(const v4f*)(src + i + 4);
    v8bf r;
    r[0] = f2bf(a[0]); r[1] = f2bf(a[1]); r[2] = f2bf(a[2]); r[3] = f2bf(a[3]);
    r[4] = f2bf(b[0]); r[5] = f2bf(b[1]); r[6] = f2bf(b[2]); r[7] = f2bf(b[3]);
    *(v8bf*)(dst + i) = r;
}

// ---------------------------------------------------------------------------
// GEMM: C[m,n] = sum_k A[m,k] * W[n,k]  (A: MxK bf16, W: NxK bf16, C: MxN f32)
// Both operands K-contiguous -> direct WMMA fragment loads.
// Wave tile 64x64 = 4x4 WMMA tiles (16 x v8f accumulators) -> 32 FLOP/byte of
// fragment traffic, 16 independent WMMAs per K-step for latency hiding.
// Block: 256 threads = 8 waves (2m x 4n), WG tile 128x256.
// ---------------------------------------------------------------------------
__global__ __launch_bounds__(256, 1)
void gemm_bf16_wmma(const __bf16* __restrict__ A, const __bf16* __restrict__ W,
                    float* __restrict__ C, int M, int N, int K) {
    const int lane  = threadIdx.x & 31;
    const int wave  = threadIdx.x >> 5;
    const int wm    = wave >> 2;          // 0..1
    const int wn    = wave & 3;           // 0..3
    const int m0    = blockIdx.y * 128 + wm * 64;
    const int n0    = blockIdx.x * 256 + wn * 64;
    const int lrow  = lane & 15;
    const int lhalf = lane >> 4;

    v8f acc[4][4];
#pragma unroll
    for (int mi = 0; mi < 4; ++mi)
#pragma unroll
        for (int ni = 0; ni < 4; ++ni) acc[mi][ni] = (v8f){};

    const size_t ldk = (size_t)K;
    // A fragment: lane holds row m; 16 contiguous K at lhalf*16
    const __bf16* ap[4];
#pragma unroll
    for (int mi = 0; mi < 4; ++mi)
        ap[mi] = A + (size_t)(m0 + mi * 16 + lrow) * ldk + (size_t)lhalf * 16;
    // B fragment: lane holds W-row n; 16 contiguous K at lhalf*16
    const __bf16* bp[4];
#pragma unroll
    for (int ni = 0; ni < 4; ++ni)
        bp[ni] = W + (size_t)(n0 + ni * 16 + lrow) * ldk + (size_t)lhalf * 16;

    for (int k = 0; k < K; k += 32) {
        v16bf a[4], b[4];
#pragma unroll
        for (int mi = 0; mi < 4; ++mi) {
            v8bf lo = *(const v8bf*)(ap[mi] + k);
            v8bf hi = *(const v8bf*)(ap[mi] + k + 8);
            a[mi] = __builtin_shufflevector(lo, hi, 0,1,2,3,4,5,6,7,8,9,10,11,12,13,14,15);
        }
#pragma unroll
        for (int ni = 0; ni < 4; ++ni)
            b[ni] = *(const v16bf*)(bp[ni] + k);

#pragma unroll
        for (int mi = 0; mi < 4; ++mi)
#pragma unroll
            for (int ni = 0; ni < 4; ++ni)
                acc[mi][ni] = __builtin_amdgcn_wmma_f32_16x16x32_bf16(
                    false, a[mi], false, b[ni], (short)0, acc[mi][ni], false, false);
    }

    // C/D layout: lane L, VGPR j -> (M = j + (L>>4)*8, N = L&15)
#pragma unroll
    for (int mi = 0; mi < 4; ++mi) {
        float* crow = C + (size_t)(m0 + mi * 16 + lhalf * 8) * N + n0 + lrow;
#pragma unroll
        for (int ni = 0; ni < 4; ++ni)
#pragma unroll
            for (int j = 0; j < 8; ++j)
                crow[(size_t)j * N + ni * 16] = acc[mi][ni][j];
    }
}

// ---------------------------------------------------------------------------
// swiglu gating: gate = sigmoid(f); v = silu(i) * (1 - gate). In-place:
// bufI: i -> v ; bufF: f -> gate. 4 elements / thread.
// ---------------------------------------------------------------------------
__global__ __launch_bounds__(256)
void swiglu_kernel(float* bufI, float* bufF, size_t n4) {
    size_t i = (size_t)blockIdx.x * blockDim.x + threadIdx.x;
    if (i >= n4) return;
    v4f iv = ((const v4f*)bufI)[i];
    v4f fv = ((const v4f*)bufF)[i];
    v4f gate, v;
#pragma unroll
    for (int j = 0; j < 4; ++j) {
        float s  = sigmoidf_(fv[j]);
        float si = iv[j] * sigmoidf_(iv[j]);
        gate[j]  = s;
        v[j]     = si * (1.0f - s);
    }
    ((v4f*)bufF)[i] = gate;
    ((v4f*)bufI)[i] = v;
}

// ---------------------------------------------------------------------------
// Chunked linear recurrence h_t = gate_t * h_{t-1} + v_t  (h_0 = 0), scan on T.
// Phase 1: per (b,chunk,d) compute a = prod(gate), h = local scan tail.
// ---------------------------------------------------------------------------
__global__ __launch_bounds__(256)
void scan_local_kernel(const float* __restrict__ gate, const float* __restrict__ v,
                       float* __restrict__ Ag, float* __restrict__ Bg,
                       int H, int T, int NC) {
    size_t tid = (size_t)blockIdx.x * blockDim.x + threadIdx.x; // (b*NC+c)*H + d
    int d  = (int)(tid % H);
    int bc = (int)(tid / H);
    int c  = bc % NC;
    int b  = bc / NC;
    int Tc = T / NC;
    size_t base = ((size_t)b * T + (size_t)c * Tc) * H + d;
    float a = 1.0f, h = 0.0f;
    for (int t = 0; t < Tc; ++t) {
        float gt = gate[base + (size_t)t * H];
        float vt = v[base + (size_t)t * H];
        a *= gt;
        h = gt * h + vt;
    }
    Ag[tid] = a;
    Bg[tid] = h;
}

// Phase 2: per (b,d) sweep carries over NC chunks (NC=64: cheap serial sweep).
__global__ __launch_bounds__(256)
void scan_carry_kernel(const float* __restrict__ Ag, const float* __restrict__ Bg,
                       float* __restrict__ Hin, int H, int NC) {
    size_t tid = (size_t)blockIdx.x * blockDim.x + threadIdx.x; // b*H + d
    int d = (int)(tid % H);
    int b = (int)(tid / H);
    float h = 0.0f;
    for (int c = 0; c < NC; ++c) {
        size_t idx = ((size_t)b * NC + c) * H + d;
        Hin[idx] = h;
        h = Ag[idx] * h + Bg[idx];
    }
}

// Phase 3: re-scan each chunk seeded with its carry; o overwrites v (bufI).
__global__ __launch_bounds__(256)
void scan_apply_kernel(const float* __restrict__ gate, float* __restrict__ v_o,
                       const float* __restrict__ Hin, int H, int T, int NC) {
    size_t tid = (size_t)blockIdx.x * blockDim.x + threadIdx.x; // (b*NC+c)*H + d
    int d  = (int)(tid % H);
    int bc = (int)(tid / H);
    int c  = bc % NC;
    int b  = bc / NC;
    int Tc = T / NC;
    size_t base = ((size_t)b * T + (size_t)c * Tc) * H + d;
    float h = Hin[tid];
    for (int t = 0; t < Tc; ++t) {
        size_t idx = base + (size_t)t * H;
        h = gate[idx] * h + v_o[idx];
        v_o[idx] = h;
    }
}

// ---------------------------------------------------------------------------
// Fused RMSNorm * w * g * sigmoid(g), output bf16 (feeds final WMMA GEMM).
// One 256-thread block (8 waves) per (b,t) row of H=2048.
// ---------------------------------------------------------------------------
__global__ __launch_bounds__(256)
void rmsnorm_gate_kernel(const float* __restrict__ o, const float* __restrict__ g,
                         const float* __restrict__ w, __bf16* __restrict__ ob, int H) {
    const size_t row = blockIdx.x;
    const float* orow = o + row * H;
    const float* grow = g + row * H;
    __bf16* obrow = ob + row * H;

    float ss = 0.0f;
    for (int i = threadIdx.x; i < H; i += 256) {
        float x = orow[i];
        ss += x * x;
    }
#pragma unroll
    for (int off = 16; off > 0; off >>= 1)
        ss += __shfl_xor(ss, off, 32);

    __shared__ float red[8];
    if ((threadIdx.x & 31) == 0) red[threadIdx.x >> 5] = ss;
    __syncthreads();
    float tot = 0.0f;
#pragma unroll
    for (int j = 0; j < 8; ++j) tot += red[j];
    const float rms = __frsqrt_rn(tot / (float)H + 1e-5f);

    for (int i = threadIdx.x; i < H; i += 256) {
        float gv  = grow[i];
        float val = orow[i] * rms * w[i] * gv * sigmoidf_(gv);
        obrow[i]  = f2bf(val);
    }
}

// ---------------------------------------------------------------------------
// Host-side orchestration
// ---------------------------------------------------------------------------
static inline size_t align_up(size_t x) { return (x + 255) & ~(size_t)255; }

extern "C" void kernel_launch(void* const* d_in, const int* in_sizes, int n_in,
                              void* d_out, int out_size, void* d_ws, size_t ws_size,
                              hipStream_t stream) {
    const float* x  = (const float*)d_in[0];
    const float* Wi = (const float*)d_in[1];
    const float* Wf = (const float*)d_in[2];
    const float* Wg = (const float*)d_in[3];
    const float* Wo = (const float*)d_in[4];
    const float* gw = (const float*)d_in[5];

    const int    H   = in_sizes[5];               // 2048
    const size_t MH  = (size_t)in_sizes[0];       // B*T*H
    const int    M   = (int)(MH / (size_t)H);     // B*T = 8192
    const int    T   = 2048;
    const int    Bb  = M / T;                     // 4
    const int    NC  = 64;                        // chunks, Tc = 32
    const size_t WSZ = (size_t)H * (size_t)H;     // weight elements

    // ---- workspace carve-up ----
    char* ws = (char*)d_ws;
    size_t off = 0;
    __bf16* xb  = (__bf16*)(ws + off); off = align_up(off + MH * 2);
    __bf16* Wib = (__bf16*)(ws + off); off = align_up(off + WSZ * 2);
    __bf16* Wfb = (__bf16*)(ws + off); off = align_up(off + WSZ * 2);
    __bf16* Wgb = (__bf16*)(ws + off); off = align_up(off + WSZ * 2);
    __bf16* Wob = (__bf16*)(ws + off); off = align_up(off + WSZ * 2);
    float*  bufI = (float*)(ws + off); off = align_up(off + MH * 4);  // i -> v -> o
    float*  bufF = (float*)(ws + off); off = align_up(off + MH * 4);  // f -> gate
    float*  bufG = (float*)(ws + off); off = align_up(off + MH * 4);  // g
    __bf16* ob   = (__bf16*)(ws + off); off = align_up(off + MH * 2); // normed gated o
    const size_t CH = (size_t)Bb * NC * H;
    float* Ag  = (float*)(ws + off); off = align_up(off + CH * 4);
    float* Bg  = (float*)(ws + off); off = align_up(off + CH * 4);
    float* Hin = (float*)(ws + off); off = align_up(off + CH * 4);
    (void)ws_size; (void)n_in; (void)out_size;

    // ---- 1) fp32 -> bf16 conversions ----
    {
        dim3 blk(256);
        dim3 gx((unsigned)((MH / 8 + 255) / 256));
        cvt_bf16_kernel<<<gx, blk, 0, stream>>>(x, xb, MH);
        dim3 gwgrid((unsigned)((WSZ / 8 + 255) / 256));
        cvt_bf16_kernel<<<gwgrid, blk, 0, stream>>>(Wi, Wib, WSZ);
        cvt_bf16_kernel<<<gwgrid, blk, 0, stream>>>(Wf, Wfb, WSZ);
        cvt_bf16_kernel<<<gwgrid, blk, 0, stream>>>(Wg, Wgb, WSZ);
        cvt_bf16_kernel<<<gwgrid, blk, 0, stream>>>(Wo, Wob, WSZ);
    }

    // ---- 2) projections i, f, g  (WMMA bf16), WG tile 128x256 ----
    {
        dim3 blk(256);
        dim3 grd((unsigned)(H / 256), (unsigned)(M / 128));
        gemm_bf16_wmma<<<grd, blk, 0, stream>>>(xb, Wib, bufI, M, H, H);
        gemm_bf16_wmma<<<grd, blk, 0, stream>>>(xb, Wfb, bufF, M, H, H);
        gemm_bf16_wmma<<<grd, blk, 0, stream>>>(xb, Wgb, bufG, M, H, H);
    }

    // ---- 3) swiglu gating (in place) ----
    {
        size_t n4 = MH / 4;
        swiglu_kernel<<<(unsigned)((n4 + 255) / 256), 256, 0, stream>>>(bufI, bufF, n4);
    }

    // ---- 4) chunked gated scan over T ----
    {
        unsigned g1 = (unsigned)((CH + 255) / 256);
        scan_local_kernel<<<g1, 256, 0, stream>>>(bufF, bufI, Ag, Bg, H, T, NC);
        unsigned g2 = (unsigned)(((size_t)Bb * H + 255) / 256);
        scan_carry_kernel<<<g2, 256, 0, stream>>>(Ag, Bg, Hin, H, NC);
        scan_apply_kernel<<<g1, 256, 0, stream>>>(bufF, bufI, Hin, H, T, NC);
    }

    // ---- 5) fused RMSNorm * w * g * sigmoid(g)  -> bf16 ----
    rmsnorm_gate_kernel<<<(unsigned)M, 256, 0, stream>>>(bufI, bufG, gw, ob, H);

    // ---- 6) output projection (WMMA bf16) -> d_out (f32) ----
    {
        dim3 blk(256);
        dim3 grd((unsigned)(H / 256), (unsigned)(M / 128));
        gemm_bf16_wmma<<<grd, blk, 0, stream>>>(ob, Wob, (float*)d_out, M, H, H);
    }
}
